// STSFullSelfAttention_45561013076444
// MI455X (gfx1250) — matrix operand order repaired and checked by reading, not compiled, not run on hardware
//
#include <hip/hip_runtime.h>
#include <hip/hip_bf16.h>

#define D_MODEL 128
#define N_HEADS 4
#define HEAD_DIM 32
#define SEQ_L 4096
#define BATCH 2
#define BH (BATCH * N_HEADS) // 8

typedef __bf16 bf16_t;
typedef __attribute__((ext_vector_type(8)))  __bf16 v8bf;
typedef __attribute__((ext_vector_type(16))) __bf16 v16bf;
typedef __attribute__((ext_vector_type(8)))  float  v8f;

__device__ __forceinline__ v8f zero8() {
  v8f r;
#pragma unroll
  for (int i = 0; i < 8; ++i) r[i] = 0.0f;
  return r;
}

__device__ __forceinline__ v16bf combine16(v8bf lo, v8bf hi) {
  v16bf r;
#pragma unroll
  for (int i = 0; i < 8; ++i) { r[i] = lo[i]; r[i + 8] = hi[i]; }
  return r;
}

__device__ __forceinline__ v16bf cvt16(const float* __restrict__ p0,
                                       const float* __restrict__ p1) {
  v16bf r;
#pragma unroll
  for (int i = 0; i < 8; ++i) { r[i] = (__bf16)p0[i]; r[i + 8] = (__bf16)p1[i]; }
  return r;
}

// CDNA5 async copy: 16 bytes global -> LDS, tracked by ASYNCcnt.
// VDST = LDS byte offset (VGPR), VADDR = 64-bit global address, GV mode.
__device__ __forceinline__ void async_copy16(unsigned lds_off, const void* gptr) {
  asm volatile("global_load_async_to_lds_b128 %0, %1, off"
               :: "v"(lds_off), "v"((unsigned long long)(uintptr_t)gptr)
               : "memory");
}
__device__ __forceinline__ void wait_async0() {
  asm volatile("s_wait_asynccnt 0x0" ::: "memory");
}

// ---------------------------------------------------------------------------
// Kernel 1: fused QKV projection.  x[8192,128] (f32) @ W^T -> bf16 buffers.
//   type 0: Q (scaled by Dh^-0.5 * log2(e))  -> qbuf [bh][L][32]
//   type 1: K                                -> kbuf [bh][L][32]
//   type 2: V                                -> vtbuf[bh][32][L] (transposed)
// ---------------------------------------------------------------------------
__global__ __launch_bounds__(128) void proj_qkv_kernel(
    const float* __restrict__ xq, const float* __restrict__ xk,
    const float* __restrict__ xv,
    const float* __restrict__ WQ, const float* __restrict__ bQ,
    const float* __restrict__ WK, const float* __restrict__ bK,
    const float* __restrict__ WV, const float* __restrict__ bV,
    bf16_t* __restrict__ qbuf, bf16_t* __restrict__ kbuf,
    bf16_t* __restrict__ vtbuf) {
  const int lane = threadIdx.x & 31;
  const int wave = threadIdx.x >> 5;
  int tile = blockIdx.x * 4 + wave;          // 0..12287
  const int type = tile / 4096;              // 0:q 1:k 2:v
  tile -= type * 4096;
  const int mt = tile >> 3;
  const int nt = tile & 7;
  const int m = lane & 15;
  const int h = lane >> 4;

  const float* X  = (type == 0) ? xq : (type == 1) ? xk : xv;
  const float* W  = (type == 0) ? WQ : (type == 1) ? WK : WV;
  const float* Bv = (type == 0) ? bQ : (type == 1) ? bK : bV;

  const float* xrow = X + (size_t)(mt * 16 + m) * D_MODEL;
  const float* wrow = W + (size_t)(nt * 16 + m) * D_MODEL;

  v8f acc = zero8();
#pragma unroll
  for (int kc = 0; kc < D_MODEL; kc += 32) {
    v16bf a = cvt16(xrow + kc + h * 8, xrow + kc + 16 + h * 8);
    v16bf b = cvt16(wrow + kc + h * 8, wrow + kc + 16 + h * 8);
    acc = __builtin_amdgcn_wmma_f32_16x16x32_bf16(false, a, false, b,
                                                  (short)0, acc, false, false);
  }

  const int j = nt * 16 + m;
  const float bias  = Bv[j];
  // Fold log2(e) into the query scale so softmax can use native exp2.
  const float scale = (type == 0) ? rsqrtf((float)HEAD_DIM) * 1.4426950408889634f
                                  : 1.0f;
  const int hh = j >> 5;
  const int d  = j & 31;
#pragma unroll
  for (int r = 0; r < 8; ++r) {
    const int row  = mt * 16 + h * 8 + r;    // x row = l*B + b
    const int l    = row / BATCH;
    const int bidx = row - l * BATCH;
    const float val = (acc[r] + bias) * scale;
    if (type == 2) {
      vtbuf[((size_t)(bidx * N_HEADS + hh) * 32 + d) * SEQ_L + l] = (bf16_t)val;
    } else {
      bf16_t* dst = (type == 0) ? qbuf : kbuf;
      dst[((size_t)(bidx * N_HEADS + hh) * SEQ_L + l) * 32 + d] = (bf16_t)val;
    }
  }
}

// ---------------------------------------------------------------------------
// Kernel 2: flash attention, block-causal (64-row block) mask.
// grid = (64 q-time-blocks, 8 b*h), block = 128 threads (4 waves).
// K/V staged to LDS once per 64-key block via async-to-LDS, double-buffered.
// Softmax state updated once per 64 keys (log2-domain, native exp2).
// ---------------------------------------------------------------------------
__global__ __launch_bounds__(128) void attn_kernel(
    const bf16_t* __restrict__ qbuf, const bf16_t* __restrict__ kbuf,
    const bf16_t* __restrict__ vtbuf, float* __restrict__ ctx) {
  __shared__ __align__(16) bf16_t kt[2][64][32];    // [buf][key][dim]  4KB each
  __shared__ __align__(16) bf16_t vt[2][32][64];    // [buf][dim][key]  4KB each
  __shared__ __align__(16) bf16_t p_lds[4][16][64]; // per-wave P tile, 8KB

  const int t    = threadIdx.x;
  const int lane = t & 31;
  const int wave = t >> 5;
  const int qt = blockIdx.x;                 // time block 0..63
  const int bh = blockIdx.y;                 // 0..7
  const int m = lane & 15;
  const int h = lane >> 4;
  const int n = lane & 15;
  const int qbase = qt * 64 + wave * 16;

  // Stage one 64-key block (K: 4KB contiguous; V^T: 32 rows x 128B) into LDS.
  auto stage = [&](int k0, int buf) {
    const char* ksrc = (const char*)(kbuf + ((size_t)bh * SEQ_L + k0) * 32);
    const unsigned kdst = (unsigned)(uintptr_t)&kt[buf][0][0];
    async_copy16(kdst + t * 16, ksrc + t * 16);
    async_copy16(kdst + 2048 + t * 16, ksrc + 2048 + t * 16);
#pragma unroll
    for (int j2 = 0; j2 < 2; ++j2) {
      const int e = t + 128 * j2;            // 256 16B transfers
      const int d = e >> 3, c = e & 7;
      const char* vsrc =
          (const char*)(vtbuf + ((size_t)bh * 32 + d) * SEQ_L + k0) + c * 16;
      const unsigned vdst = (unsigned)(uintptr_t)&vt[buf][d][0] + c * 16;
      async_copy16(vdst, vsrc);
    }
  };

  // Q tile in A-operand layout (pre-scaled bf16)
  const bf16_t* qb = qbuf + ((size_t)bh * SEQ_L + qbase + m) * 32;
  const v16bf qa = combine16(*(const v8bf*)(qb + h * 16),
                             *(const v8bf*)(qb + h * 16 + 8));

  v8f O0 = zero8(), O1 = zero8();
  float mrow[8], lrow[8];
#pragma unroll
  for (int r = 0; r < 8; ++r) { mrow[r] = -__builtin_inff(); lrow[r] = 0.0f; }

  const int nkb = qt + 1;                    // 64-key blocks, fully unmasked
  stage(0, 0);
  wait_async0();
  __syncthreads();

  for (int kb = 0; kb < nkb; ++kb) {
    const int buf = kb & 1;
    if (kb + 1 < nkb) stage((kb + 1) * 64, buf ^ 1); // overlap with compute

    // ---- scores: 4 x (16x16x32) WMMA over 64 keys, from LDS ----
    const v16bf B0 = *(const v16bf*)&kt[buf][n][h * 16];
    const v16bf B1 = *(const v16bf*)&kt[buf][16 + n][h * 16];
    const v16bf B2 = *(const v16bf*)&kt[buf][32 + n][h * 16];
    const v16bf B3 = *(const v16bf*)&kt[buf][48 + n][h * 16];
    v8f S0 = __builtin_amdgcn_wmma_f32_16x16x32_bf16(false, qa, false, B0,
                                                     (short)0, zero8(), false, false);
    v8f S1 = __builtin_amdgcn_wmma_f32_16x16x32_bf16(false, qa, false, B1,
                                                     (short)0, zero8(), false, false);
    v8f S2 = __builtin_amdgcn_wmma_f32_16x16x32_bf16(false, qa, false, B2,
                                                     (short)0, zero8(), false, false);
    v8f S3 = __builtin_amdgcn_wmma_f32_16x16x32_bf16(false, qa, false, B3,
                                                     (short)0, zero8(), false, false);

    // ---- online softmax (log2 domain), one update per 64 keys ----
#pragma unroll
    for (int r = 0; r < 8; ++r) {
      float mx = fmaxf(fmaxf(S0[r], S1[r]), fmaxf(S2[r], S3[r]));
      for (int mk = 1; mk < 16; mk <<= 1) mx = fmaxf(mx, __shfl_xor(mx, mk, 32));
      const float mnew  = fmaxf(mrow[r], mx);
      const float alpha = exp2f(mrow[r] - mnew);
      const float p0 = exp2f(S0[r] - mnew);
      const float p1 = exp2f(S1[r] - mnew);
      const float p2 = exp2f(S2[r] - mnew);
      const float p3 = exp2f(S3[r] - mnew);
      float s = (p0 + p1) + (p2 + p3);
      for (int mk = 1; mk < 16; mk <<= 1) s += __shfl_xor(s, mk, 32);
      lrow[r] = lrow[r] * alpha + s;
      mrow[r] = mnew;
      O0[r] *= alpha;
      O1[r] *= alpha;
      p_lds[wave][h * 8 + r][n]      = (bf16_t)p0;
      p_lds[wave][h * 8 + r][16 + n] = (bf16_t)p1;
      p_lds[wave][h * 8 + r][32 + n] = (bf16_t)p2;
      p_lds[wave][h * 8 + r][48 + n] = (bf16_t)p3;
    }
    // Per-wave LDS round-trip: same-wave LDS ops are in order; just stop the
    // compiler from reordering the reads above the writes.
    asm volatile("" ::: "memory");

    // ---- P·V: 4 x (16x16x32) WMMA ----
#pragma unroll
    for (int sub = 0; sub < 2; ++sub) {
      const v16bf Pa =
          combine16(*(const v8bf*)&p_lds[wave][m][sub * 32 + h * 16],
                    *(const v8bf*)&p_lds[wave][m][sub * 32 + h * 16 + 8]);
      const v16bf V0 = *(const v16bf*)&vt[buf][n][sub * 32 + h * 16];
      const v16bf V1 = *(const v16bf*)&vt[buf][16 + n][sub * 32 + h * 16];
      O0 = __builtin_amdgcn_wmma_f32_16x16x32_bf16(false, Pa, false, V0,
                                                   (short)0, O0, false, false);
      O1 = __builtin_amdgcn_wmma_f32_16x16x32_bf16(false, Pa, false, V1,
                                                   (short)0, O1, false, false);
    }

    wait_async0();      // next block fully staged (no-op on last iteration)
    __syncthreads();    // all waves done with kt/vt[buf] before reuse
  }

#pragma unroll
  for (int r = 0; r < 8; ++r) {
    const float inv = 1.0f / lrow[r];
    const int row = qbase + h * 8 + r;
    ctx[((size_t)bh * SEQ_L + row) * 32 + n]      = O0[r] * inv;
    ctx[((size_t)bh * SEQ_L + row) * 32 + 16 + n] = O1[r] * inv;
  }
}

// ---------------------------------------------------------------------------
// Kernel 3: output projection. out[b,l,:] = concat_heads(ctx) @ W_O^T + b_O.
// ---------------------------------------------------------------------------
__global__ __launch_bounds__(128) void outproj_kernel(
    const float* __restrict__ ctx, const float* __restrict__ WO,
    const float* __restrict__ bO, float* __restrict__ out) {
  const int lane = threadIdx.x & 31;
  const int wave = threadIdx.x >> 5;
  const int tile = blockIdx.x * 4 + wave;    // 0..4095
  const int mt = tile >> 3;
  const int nt = tile & 7;
  const int m = lane & 15;
  const int h = lane >> 4;

  const int row  = mt * 16 + m;              // = b*L + l
  const int bidx = row >> 12;
  const int l    = row & (SEQ_L - 1);
  const float* wrow = WO + (size_t)(nt * 16 + m) * D_MODEL;

  v8f acc = zero8();
#pragma unroll
  for (int hh = 0; hh < N_HEADS; ++hh) {
    const float* crow = ctx + ((size_t)(bidx * N_HEADS + hh) * SEQ_L + l) * 32;
    v16bf a = cvt16(crow + h * 8, crow + 16 + h * 8);
    v16bf b = cvt16(wrow + hh * 32 + h * 8, wrow + hh * 32 + 16 + h * 8);
    acc = __builtin_amdgcn_wmma_f32_16x16x32_bf16(false, a, false, b,
                                                  (short)0, acc, false, false);
  }

  const float bias = bO[nt * 16 + m];
#pragma unroll
  for (int r = 0; r < 8; ++r) {
    const int rr = mt * 16 + h * 8 + r;
    out[(size_t)rr * D_MODEL + nt * 16 + m] = acc[r] + bias;
  }
}

// ---------------------------------------------------------------------------
extern "C" void kernel_launch(void* const* d_in, const int* in_sizes, int n_in,
                              void* d_out, int out_size, void* d_ws,
                              size_t ws_size, hipStream_t stream) {
  (void)in_sizes; (void)n_in; (void)out_size; (void)ws_size;
  const float* query = (const float*)d_in[0];
  const float* key   = (const float*)d_in[1];
  const float* value = (const float*)d_in[2];
  const float* WQ = (const float*)d_in[3];
  const float* bQ = (const float*)d_in[4];
  const float* WK = (const float*)d_in[5];
  const float* bK = (const float*)d_in[6];
  const float* WV = (const float*)d_in[7];
  const float* bV = (const float*)d_in[8];
  const float* WO = (const float*)d_in[9];
  const float* bO = (const float*)d_in[10];
  float* out = (float*)d_out;

  // Workspace layout (10 MiB total):
  //   [0,2Mi)  qbuf  bf16 [bh][L][32]
  //   [2,4Mi)  kbuf  bf16 [bh][L][32]
  //   [4,6Mi)  vtbuf bf16 [bh][32][L]
  //   [6,10Mi) ctx   f32  [bh][L][32]
  char* ws = (char*)d_ws;
  bf16_t* qbuf  = (bf16_t*)(ws);
  bf16_t* kbuf  = (bf16_t*)(ws + (2u << 20));
  bf16_t* vtbuf = (bf16_t*)(ws + (4u << 20));
  float*  ctx   = (float*)(ws + (6u << 20));

  proj_qkv_kernel<<<3072, 128, 0, stream>>>(query, key, value, WQ, bQ, WK, bK,
                                            WV, bV, qbuf, kbuf, vtbuf);
  attn_kernel<<<dim3(64, 8), 128, 0, stream>>>(qbuf, kbuf, vtbuf, ctx);
  outproj_kernel<<<1024, 128, 0, stream>>>(ctx, WO, bO, out);
}
